// NeuralSDE_35871566856363
// MI455X (gfx1250) — matrix operand order repaired and checked by reading, not compiled, not run on hardware
//
#include <hip/hip_runtime.h>
#include <hip/hip_bf16.h>

typedef __attribute__((ext_vector_type(16))) _Float16 v16h;
typedef __attribute__((ext_vector_type(8)))  _Float16 v8h;
typedef __attribute__((ext_vector_type(8)))  float    v8f;

#define N_STEPS 96
#define N_MAT   4
#define PERIOD  24
#define NSTR    201
#define BATCH   8192
#define PATHS   32                  // paths per block (one per thread, ONE wave)
#define NBLK    (BATCH / PATHS)     // 256 blocks
#define KT_TOT  13                  // ceil(201/16)

#define RFR   0.05f
#define S0C   100.0f
#define DTC   (2.0f / 96.0f)
#define SQDT  0.14433756729740643f  // sqrt(DT)

// ---- LDS layout (dynamic shared) ----
// halves: Xh[32*32] | bufA[32*160] | bufB[32*160] | Gh[4*32*64]
//       | dW0s[96*32] | dW1s[96*32] | dW2s[96*32]
//       | hW0s[64*32] | hW1s[64*64] | hW2s[64*64] | hW3s[64*64]
// floats: biasH[256] | biasD[288] | csumh[128] | Sarr[32] | W3l[96] | b3l[4]
//       | lsum[208] | lsq[208]
#define SMEM_HALF  (32*32 + 2*32*160 + 4*32*64 + 3*96*32 + 64*32 + 3*64*64) // 43008
#define SMEM_FLOAT (256 + 288 + 128 + 32 + 96 + 4 + 208 + 208)             // 1220
#define SMEM_BYTES (SMEM_HALF*2 + SMEM_FLOAT*4)                            // 90896

__device__ inline v8f wmma_f16(v16h a, v16h b, v8f c) {
  // D = A(16x32 f16) * B(32x16 f16) + C(16x16 f32)
  return __builtin_amdgcn_wmma_f32_16x16x32_f16(false, a, false, b, (short)0, c,
                                                false, false);
}

// A-fragment (16x32 f16) from row-major f16 weights staged in LDS.
// Per lane: 8 halves at K=kb..kb+7 and 8 at K=kb+16..kb+23 -> 2x ds_load_b128.
__device__ inline v16h a_frag_lds(const _Float16* W, int ldh, int mtile, int kh,
                                  int lane) {
  int mr = mtile * 16 + (lane & 15);
  int kb = ((lane & 16) ? 8 : 0) + kh * 32;
  const _Float16* row = W + (size_t)mr * ldh + kb;
  v8h lo = *(const v8h*)(row);
  v8h hi = *(const v8h*)(row + 16);
  return __builtin_shufflevector(lo, hi, 0, 1, 2, 3, 4, 5, 6, 7,
                                 8, 9, 10, 11, 12, 13, 14, 15);
}

// A-fragment from hW4[p] (global f32) gathering rows k*4+mat, OOB rows -> 0.
__device__ inline v16h a_frag_hw4(const float* __restrict__ hW4p, int kt,
                                  int mat, int kh, int lane) {
  int krow = kt * 16 + (lane & 15);
  bool valid = (krow < NSTR);
  const float* row = hW4p + (size_t)(min(krow, NSTR - 1) * 4 + mat) * 64;
  int kb = ((lane & 16) ? 8 : 0) + kh * 32;
  v16h a;
#pragma unroll
  for (int i = 0; i < 8; ++i) {
    int k = kb + ((i < 4) ? 2 * i : 16 + 2 * (i - 4));
    a[2 * i]     = (_Float16)(valid ? row[k]     : 0.0f);
    a[2 * i + 1] = (_Float16)(valid ? row[k + 1] : 0.0f);
  }
  return a;
}

// B-fragment (32x16 f16) from path-major LDS activations: src[path][feature].
__device__ inline v16h b_frag(const _Float16* src, int ldh, int nbase, int kofs,
                              int lane) {
  int n = nbase + (lane & 15);
  int kb = (lane & 16) ? 8 : 0;
  const _Float16* p = src + (size_t)n * ldh + kofs + kb;
  v8h lo = *(const v8h*)(p);
  v8h hi = *(const v8h*)(p + 16);
  return __builtin_shufflevector(lo, hi, 0, 1, 2, 3, 4, 5, 6, 7,
                                 8, 9, 10, 11, 12, 13, 14, 15);
}

// Per-lane 8-row bias vector (C-matrix initializer) from an LDS bias row array.
__device__ inline v8f bias_vec(const float* barr, int fbase, int lane) {
  return *(const v8f*)(barr + fbase + ((lane & 16) ? 8 : 0));
}

// Store D (bias pre-accumulated) with ReLU (v_med3) as f16, path-major.
__device__ inline void d_store_relu(_Float16* dst, int ldh, int nbase, int fbase,
                                    v8f acc, int lane) {
  int n  = nbase + (lane & 15);
  int f0 = fbase + ((lane & 16) ? 8 : 0);
  _Float16* p = dst + (size_t)n * ldh + f0;
  v8f t;
#pragma unroll
  for (int r = 0; r < 8; ++r)
    t[r] = __builtin_amdgcn_fmed3f(acc[r], 0.0f, __builtin_inff());  // relu
  *reinterpret_cast<v8h*>(p) = __builtin_convertvector(t, v8h);      // 16B store
}

__global__ __launch_bounds__(PATHS, 1)
void sde_sim_kernel(const float* __restrict__ nW0, const float* __restrict__ nb0,
                    const float* __restrict__ nW1, const float* __restrict__ nb1,
                    const float* __restrict__ nW2, const float* __restrict__ nb2,
                    const float* __restrict__ nW3, const float* __restrict__ nb3,
                    const float* __restrict__ hW0, const float* __restrict__ hb0,
                    const float* __restrict__ hW1, const float* __restrict__ hb1,
                    const float* __restrict__ hW2, const float* __restrict__ hb2,
                    const float* __restrict__ hW3, const float* __restrict__ hb3,
                    const float* __restrict__ hW4, const float* __restrict__ hb4,
                    const float* __restrict__ rho, const float* __restrict__ V0,
                    const float* __restrict__ strikes,
                    const float* __restrict__ noise1,
                    const float* __restrict__ noise2,
                    float* __restrict__ psum, float* __restrict__ psq)
{
  const int tid   = threadIdx.x;          // == lane (single wave)
  const int lane  = tid;
  const int blk   = blockIdx.x;
  const int gpath = blk * PATHS + tid;

  extern __shared__ char smem[];
  _Float16* Xh   = (_Float16*)smem;        // 32 x 32  (t,S,V,0...)
  _Float16* bufA = Xh   + 32 * 32;         // 32 x 160 (drift 0..95, hedge 96..159)
  _Float16* bufB = bufA + 32 * 160;
  _Float16* Gh   = bufB + 32 * 160;        // 4 x 32 x 64 (per-period G history)
  _Float16* dW0s = Gh   + 4 * 32 * 64;     // 96 x 32 (K padded 3->32)
  _Float16* dW1s = dW0s + 96 * 32;         // 96 x 32
  _Float16* dW2s = dW1s + 96 * 32;         // 96 x 32
  _Float16* hW0s = dW2s + 96 * 32;         // 64 x 32 (K padded 2->32)
  _Float16* hW1s = hW0s + 64 * 32;         // 64 x 64
  _Float16* hW2s = hW1s + 64 * 64;         // 64 x 64
  _Float16* hW3s = hW2s + 64 * 64;         // 64 x 64
  float* biasH = (float*)(hW3s + 64 * 64); // 4 x 64  hedge biases
  float* biasD = biasH + 256;              // 3 x 96  drift biases
  float* csumh = biasD + 288;              // 4 x 32
  float* Sarr  = csumh + 128;              // 32
  float* W3l   = Sarr + 32;                // 96
  float* b3l   = W3l + 96;                 // 4
  float* lsum  = b3l + 4;                  // 208
  float* lsq   = lsum + 208;               // 208
  float* payB  = (float*)bufA;             // alias: 64 x 32 f32 payoff chunk

  // zero the padded K-columns of the input staging buffer (once)
  for (int i = tid; i < PATHS * 32; i += PATHS) Xh[i] = (_Float16)0.0f;

  const float rho_s = rho[0];
  const float rc = sqrtf(fmaxf(1.0f - rho_s * rho_s, 0.0f));
  const float DF = __expf(-RFR * DTC);

  float S = S0C;
  float V = V0[0];
  v8f Gv[8];
  {
    v8f z = {};
#pragma unroll
    for (int q = 0; q < 8; ++q) Gv[q] = z;
  }
  float csum = 0.0f;

  for (int m = 0; m < N_MAT; ++m) {
    // ------- period setup: stage f16 weights + biases into LDS -------
    for (int i = tid; i < 96 * 32; i += PATHS) {   // drift W0 (pad K 3->32)
      int f = i >> 5, k = i & 31;
      int net = f >> 5, o = f & 31;
      float w = (k < 3) ? nW0[((net * 4 + m) * 32 + o) * 3 + k] : 0.0f;
      dW0s[i] = (_Float16)w;
    }
    for (int i = tid; i < 96 * 32; i += PATHS) {   // drift W1, W2
      int f = i >> 5, k = i & 31;
      int net = f >> 5, o = f & 31;
      size_t gi = (size_t)((net * 4 + m) * 32 + o) * 32 + k;
      dW1s[i] = (_Float16)nW1[gi];
      dW2s[i] = (_Float16)nW2[gi];
    }
    for (int i = tid; i < 64 * 32; i += PATHS) {   // hedge W0 (pad K 2->32)
      int f = i >> 5, k = i & 31;
      float w = (k < 2) ? hW0[(m * 64 + f) * 2 + k] : 0.0f;
      hW0s[i] = (_Float16)w;
    }
    for (int i = tid; i < 64 * 64; i += PATHS) {   // hedge W1..W3
      int f = i >> 6, k = i & 63;
      size_t gi = ((size_t)m * 64 + f) * 64 + k;
      hW1s[i] = (_Float16)hW1[gi];
      hW2s[i] = (_Float16)hW2[gi];
      hW3s[i] = (_Float16)hW3[gi];
    }
    for (int i = tid; i < 4 * 64; i += PATHS) {    // hedge biases
      int l = i >> 6, f = i & 63;
      const float* src = (l == 0) ? hb0 : (l == 1) ? hb1 : (l == 2) ? hb2 : hb3;
      biasH[i] = src[m * 64 + f];
    }
    for (int i = tid; i < 3 * 96; i += PATHS) {    // drift biases
      int l = i / 96, f = i % 96;
      const float* src = (l == 0) ? nb0 : (l == 1) ? nb1 : nb2;
      biasD[i] = src[((f >> 5) * 4 + m) * 32 + (f & 31)];
    }
    for (int i = tid; i < 96; i += PATHS)          // drift output head
      W3l[i] = nW3[((i >> 5) * 4 + m) * 32 + (i & 31)];
    if (tid < 3) b3l[tid] = nb3[tid * 4 + m];
    __syncthreads();   // single-wave workgroup: effectively free

    // --------------------------- time steps ---------------------------
    for (int i = 0; i < PERIOD; ++i) {
      const int g = m * PERIOD + i;
      const float t  = (float)g * DTC;
      const float n1 = noise1[(size_t)g * BATCH + gpath];
      const float n2 = noise2[(size_t)g * BATCH + gpath];
      if (g + 1 < N_STEPS) {
        __builtin_prefetch(&noise1[(size_t)(g + 1) * BATCH + gpath], 0, 1);
        __builtin_prefetch(&noise2[(size_t)(g + 1) * BATCH + gpath], 0, 1);
      }

      // phase A: stage [t,S,V]
      {
        _Float16* x = Xh + (size_t)tid * 32;
        x[0] = (_Float16)t; x[1] = (_Float16)S; x[2] = (_Float16)V;
      }
      __syncthreads();

      // phase B: layer 0 (Xh -> bufA); B-frags shared by all 10 M-tiles
      {
        v16h b0 = b_frag(Xh, 32, 0,  0, lane);
        v16h b1 = b_frag(Xh, 32, 16, 0, lane);
#pragma unroll
        for (int st = 0; st < 6; ++st) {           // drift: 6 super-tiles
          v16h a = a_frag_lds(dW0s, 32, st, 0, lane);
          v8f bv = bias_vec(biasD + 0 * 96, st * 16, lane);
          v8f a0 = wmma_f16(a, b0, bv);
          v8f a1 = wmma_f16(a, b1, bv);
          d_store_relu(bufA, 160, 0,  st * 16, a0, lane);
          d_store_relu(bufA, 160, 16, st * 16, a1, lane);
        }
#pragma unroll
        for (int mt = 0; mt < 4; ++mt) {           // hedge: 4 M-tiles
          v16h a = a_frag_lds(hW0s, 32, mt, 0, lane);
          v8f bv = bias_vec(biasH + 0 * 64, mt * 16, lane);
          v8f a0 = wmma_f16(a, b0, bv);
          v8f a1 = wmma_f16(a, b1, bv);
          d_store_relu(bufA, 160, 0,  96 + mt * 16, a0, lane);
          d_store_relu(bufA, 160, 16, 96 + mt * 16, a1, lane);
        }
      }
      __syncthreads();

      // phase C: layer 1 (bufA -> bufB)
#pragma unroll
      for (int net = 0; net < 3; ++net) {
        v16h b0 = b_frag(bufA, 160, 0,  net * 32, lane);
        v16h b1 = b_frag(bufA, 160, 16, net * 32, lane);
#pragma unroll
        for (int half = 0; half < 2; ++half) {
          int st = net * 2 + half;
          v16h a = a_frag_lds(dW1s, 32, st, 0, lane);
          v8f bv = bias_vec(biasD + 1 * 96, st * 16, lane);
          v8f a0 = wmma_f16(a, b0, bv);
          v8f a1 = wmma_f16(a, b1, bv);
          d_store_relu(bufB, 160, 0,  st * 16, a0, lane);
          d_store_relu(bufB, 160, 16, st * 16, a1, lane);
        }
      }
      {
        v16h b00 = b_frag(bufA, 160, 0,  96,      lane);
        v16h b01 = b_frag(bufA, 160, 0,  96 + 32, lane);
        v16h b10 = b_frag(bufA, 160, 16, 96,      lane);
        v16h b11 = b_frag(bufA, 160, 16, 96 + 32, lane);
#pragma unroll
        for (int mt = 0; mt < 4; ++mt) {
          v16h aA = a_frag_lds(hW1s, 64, mt, 0, lane);
          v16h aB = a_frag_lds(hW1s, 64, mt, 1, lane);
          v8f bv = bias_vec(biasH + 1 * 64, mt * 16, lane);
          v8f a0 = wmma_f16(aA, b00, bv); a0 = wmma_f16(aB, b01, a0);
          v8f a1 = wmma_f16(aA, b10, bv); a1 = wmma_f16(aB, b11, a1);
          d_store_relu(bufB, 160, 0,  96 + mt * 16, a0, lane);
          d_store_relu(bufB, 160, 16, 96 + mt * 16, a1, lane);
        }
      }
      __syncthreads();

      // phase D: layer 2 (bufB -> bufA)
#pragma unroll
      for (int net = 0; net < 3; ++net) {
        v16h b0 = b_frag(bufB, 160, 0,  net * 32, lane);
        v16h b1 = b_frag(bufB, 160, 16, net * 32, lane);
#pragma unroll
        for (int half = 0; half < 2; ++half) {
          int st = net * 2 + half;
          v16h a = a_frag_lds(dW2s, 32, st, 0, lane);
          v8f bv = bias_vec(biasD + 2 * 96, st * 16, lane);
          v8f a0 = wmma_f16(a, b0, bv);
          v8f a1 = wmma_f16(a, b1, bv);
          d_store_relu(bufA, 160, 0,  st * 16, a0, lane);
          d_store_relu(bufA, 160, 16, st * 16, a1, lane);
        }
      }
      {
        v16h b00 = b_frag(bufB, 160, 0,  96,      lane);
        v16h b01 = b_frag(bufB, 160, 0,  96 + 32, lane);
        v16h b10 = b_frag(bufB, 160, 16, 96,      lane);
        v16h b11 = b_frag(bufB, 160, 16, 96 + 32, lane);
#pragma unroll
        for (int mt = 0; mt < 4; ++mt) {
          v16h aA = a_frag_lds(hW2s, 64, mt, 0, lane);
          v16h aB = a_frag_lds(hW2s, 64, mt, 1, lane);
          v8f bv = bias_vec(biasH + 2 * 64, mt * 16, lane);
          v8f a0 = wmma_f16(aA, b00, bv); a0 = wmma_f16(aB, b01, a0);
          v8f a1 = wmma_f16(aA, b10, bv); a1 = wmma_f16(aB, b11, a1);
          d_store_relu(bufA, 160, 0,  96 + mt * 16, a0, lane);
          d_store_relu(bufA, 160, 16, 96 + mt * 16, a1, lane);
        }
      }
      __syncthreads();

      // phase E: hedge layer 3 (bufA -> bufB) + per-thread drift head + SDE
      {
        v16h b00 = b_frag(bufA, 160, 0,  96,      lane);
        v16h b01 = b_frag(bufA, 160, 0,  96 + 32, lane);
        v16h b10 = b_frag(bufA, 160, 16, 96,      lane);
        v16h b11 = b_frag(bufA, 160, 16, 96 + 32, lane);
#pragma unroll
        for (int mt = 0; mt < 4; ++mt) {
          v16h aA = a_frag_lds(hW3s, 64, mt, 0, lane);
          v16h aB = a_frag_lds(hW3s, 64, mt, 1, lane);
          v8f bv = bias_vec(biasH + 3 * 64, mt * 16, lane);
          v8f a0 = wmma_f16(aA, b00, bv); a0 = wmma_f16(aB, b01, a0);
          v8f a1 = wmma_f16(aA, b10, bv); a1 = wmma_f16(aB, b11, a1);
          d_store_relu(bufB, 160, 0,  96 + mt * 16, a0, lane);
          d_store_relu(bufB, 160, 16, 96 + mt * 16, a1, lane);
        }
      }
      float Snew, Vnew, c;
      {
        const v8h* h2v = (const v8h*)(bufA + (size_t)tid * 160);  // drift h2
        float raw[3];
#pragma unroll
        for (int nn = 0; nn < 3; ++nn) {
          float a = b3l[nn];
#pragma unroll
          for (int q = 0; q < 4; ++q) {
            v8f hf = __builtin_convertvector(h2v[nn * 4 + q], v8f);
            v8f wf = *(const v8f*)(W3l + nn * 32 + q * 8);
#pragma unroll
            for (int r = 0; r < 8; ++r) a += wf[r] * hf[r];
          }
          raw[nn] = a;
        }
        float sigS = raw[0] / (1.0f + fabsf(raw[0]) * SQDT);
        float bV   = raw[1] / (1.0f + fabsf(raw[1]) * SQDT);
        float sigV = raw[2] / (1.0f + fabsf(raw[2]) * SQDT);
        float bS   = S * RFR / (1.0f + fabsf(S * RFR) * SQDT);
        float dW = SQDT * n2;
        float dB = rho_s * dW + rc * SQDT * n1;
        Snew = S + bS * DTC + sigS * dB;
        Vnew = fmaxf(V + bV * DTC + sigV * dW, 0.0f);
        c = DF * S * sigS * dW;   // hedge increment coefficient
      }
      __syncthreads();

      // phase F: fold hedge output layer: G += c * h3
      {
        const v8h* h3v = (const v8h*)(bufB + (size_t)tid * 160 + 96);
#pragma unroll
        for (int q = 0; q < 8; ++q) {
          v8f hf = __builtin_convertvector(h3v[q], v8f);
#pragma unroll
          for (int r = 0; r < 8; ++r) Gv[q][r] += c * hf[r];
        }
        csum += c;
      }
      S = Snew; V = Vnew;
    }

    // ------------- period end: G history + maturity-m statistics -------------
    {
      _Float16* gdst = Gh + ((size_t)m * PATHS + tid) * 64;
#pragma unroll
      for (int q = 0; q < 8; ++q)
        *reinterpret_cast<v8h*>(gdst + q * 8) = __builtin_convertvector(Gv[q], v8h);
      csumh[m * PATHS + tid] = csum;
      Sarr[tid] = S;
    }
    __syncthreads();

    const float disc = __expf(-RFR * DTC * (float)(PERIOD * (m + 1)));

    // H[k*4+m, b] = sum_{p<=m} hW4[p][k*4+m,:] @ G_p[:,b] + hb4[p][k*4+m]*csum_p[b]
    for (int kc = 0; kc < 4; ++kc) {               // chunks of 4 k-tiles
      int ktlo = kc * 4;
      int kthi = min(ktlo + 4, KT_TOT);
      for (int kt = ktlo; kt < kthi; ++kt) {
#pragma unroll
        for (int nt = 0; nt < 2; ++nt) {
          v8f acc = {};
          for (int p = 0; p <= m; ++p) {
#pragma unroll
            for (int kh = 0; kh < 2; ++kh) {
              v16h a = a_frag_hw4(hW4 + (size_t)p * 804 * 64, kt, m, kh, lane);
              v16h b = b_frag(Gh + (size_t)p * PATHS * 64, 64, nt * 16, kh * 32,
                              lane);
              acc = wmma_f16(a, b, acc);
            }
          }
          int path = nt * 16 + (lane & 15);
          float Sp = Sarr[path];
          int mrow0 = (lane & 16) ? 8 : 0;
#pragma unroll
          for (int r = 0; r < 8; ++r) {
            int krow = kt * 16 + mrow0 + r;
            int kcl = min(krow, NSTR - 1);
            float bsum = 0.0f;
            for (int p = 0; p <= m; ++p)
              bsum += hb4[p * 804 + kcl * 4 + m] * csumh[p * PATHS + path];
            float Kst = strikes[kcl];
            float pay = disc * fmaxf(Sp - Kst, 0.0f) - (acc[r] + bsum);
            payB[((kt - ktlo) * 16 + mrow0 + r) * PATHS + path] = pay;
          }
        }
      }
      __syncthreads();
      int nrows = (kthi - ktlo) * 16;
      for (int rl = tid; rl < nrows; rl += PATHS) {  // deterministic reduction
        int krow = ktlo * 16 + rl;
        if (krow < NSTR) {
          float s = 0.0f, q = 0.0f;
          for (int b = 0; b < PATHS; ++b) {
            float v = payB[rl * PATHS + b];
            s += v; q += v * v;
          }
          lsum[krow] = s; lsq[krow] = q;
        }
      }
      __syncthreads();
    }

    // flush deterministic per-block partials
    for (int k = tid; k < NSTR; k += PATHS) {
      int row = k * 4 + m;
      psum[(size_t)row * NBLK + blk] = lsum[k];
      psq [(size_t)row * NBLK + blk] = lsq[k];
    }
    {
      v8f z = {};
#pragma unroll
      for (int q = 0; q < 8; ++q) Gv[q] = z;
    }
    csum = 0.0f;
    __syncthreads();
  }
}

__global__ void sde_final_kernel(const float* __restrict__ psum,
                                 const float* __restrict__ psq,
                                 float* __restrict__ out)
{
  int r = blockIdx.x * blockDim.x + threadIdx.x;
  if (r >= NSTR * N_MAT) return;
  float s = 0.0f, q = 0.0f;
  for (int b = 0; b < NBLK; ++b) {
    s += psum[(size_t)r * NBLK + b];
    q += psq [(size_t)r * NBLK + b];
  }
  float mean = s / (float)BATCH;
  float var  = (q - s * s / (float)BATCH) / (float)(BATCH - 1);
  out[r] = mean;                 // prices (201,4) flat
  out[NSTR * N_MAT + r] = var;   // variance (201,4) flat
}

extern "C" void kernel_launch(void* const* d_in, const int* in_sizes, int n_in,
                              void* d_out, int out_size, void* d_ws, size_t ws_size,
                              hipStream_t stream) {
  (void)in_sizes; (void)n_in; (void)out_size; (void)ws_size;
  const float* nW0 = (const float*)d_in[0];
  const float* nb0 = (const float*)d_in[1];
  const float* nW1 = (const float*)d_in[2];
  const float* nb1 = (const float*)d_in[3];
  const float* nW2 = (const float*)d_in[4];
  const float* nb2 = (const float*)d_in[5];
  const float* nW3 = (const float*)d_in[6];
  const float* nb3 = (const float*)d_in[7];
  const float* hW0 = (const float*)d_in[8];
  const float* hb0 = (const float*)d_in[9];
  const float* hW1 = (const float*)d_in[10];
  const float* hb1 = (const float*)d_in[11];
  const float* hW2 = (const float*)d_in[12];
  const float* hb2 = (const float*)d_in[13];
  const float* hW3 = (const float*)d_in[14];
  const float* hb3 = (const float*)d_in[15];
  const float* hW4 = (const float*)d_in[16];
  const float* hb4 = (const float*)d_in[17];
  const float* rho = (const float*)d_in[18];
  const float* V0  = (const float*)d_in[19];
  const float* strikes = (const float*)d_in[20];
  const float* noise1  = (const float*)d_in[21];
  const float* noise2  = (const float*)d_in[22];

  float* psum = (float*)d_ws;                         // 804*256 f32
  float* psq  = psum + (size_t)NSTR * N_MAT * NBLK;   // 804*256 f32

  (void)hipFuncSetAttribute((const void*)sde_sim_kernel,
                            hipFuncAttributeMaxDynamicSharedMemorySize,
                            SMEM_BYTES);

  sde_sim_kernel<<<NBLK, PATHS, SMEM_BYTES, stream>>>(
      nW0, nb0, nW1, nb1, nW2, nb2, nW3, nb3,
      hW0, hb0, hW1, hb1, hW2, hb2, hW3, hb3, hW4, hb4,
      rho, V0, strikes, noise1, noise2, psum, psq);

  sde_final_kernel<<<(NSTR * N_MAT + 127) / 128, 128, 0, stream>>>(
      psum, psq, (float*)d_out);
}
